// VQ_Quantizer_15255723836213
// MI455X (gfx1250) — matrix operand order repaired and checked by reading, not compiled, not run on hardware
//
#include <hip/hip_runtime.h>
#include <hip/hip_bf16.h>

// ---------------------------------------------------------------------------
// VQ quantizer forward for MI455X (gfx1250, wave32, WMMA).
// S = X @ W^T via split-precision bf16x3 WMMA (v_wmma_f32_16x16x32_bf16),
// argmin_k(||w_k||^2 - 2 S), then fp32 gather + loss (exact vs reference).
// ---------------------------------------------------------------------------

typedef __bf16 bf16_t;
typedef __attribute__((ext_vector_type(16))) __bf16 v16bf;
typedef __attribute__((ext_vector_type(8)))  __bf16 v8bf;
typedef __attribute__((ext_vector_type(8)))  float  v8f;
typedef __attribute__((ext_vector_type(4)))  float  v4f;

#define NUM_EMBED   512
#define DIM_EMBED   256
#define BATCH       64
#define T_SZ        2048
#define N_ROWS      (BATCH * T_SZ)                 // 131072
#define Q_ELEMS     (BATCH * DIM_EMBED * T_SZ)     // 33554432
#define COMMIT_LOSS 0.25f

#define THREADS   256      // 8 wave32 waves
#define M_TILE    64       // rows per workgroup
#define KC        64       // codes per chunk (8 chunks)
#define A_STRIDE  264      // bf16 elems per LDS row (256 + 8 pad, 16B aligned)
#define B_STRIDE  264
#define S_STRIDE  68       // f32 elems per S row (64 + 4 pad)

// scratch layout (bytes)
#define WS_WHI      0
#define WS_WLO      (NUM_EMBED * DIM_EMBED * 2)                  // 262144
#define WS_WN2      (WS_WLO + NUM_EMBED * DIM_EMBED * 2)         // 524288
#define WS_IDX      (WS_WN2 + NUM_EMBED * 4)                     // 526336
#define WS_PART     (WS_IDX + N_ROWS * 4)                        // 1050624
#define N_PART      (Q_ELEMS / 4 / THREADS)                      // 32768

// -------------------- kernel 0: prep codebook ------------------------------
__global__ __launch_bounds__(DIM_EMBED)
void vq_prep_w(const float* __restrict__ W, bf16_t* __restrict__ Whi,
               bf16_t* __restrict__ Wlo, float* __restrict__ wn2) {
    __shared__ float red[DIM_EMBED];
    const int k = blockIdx.x, d = threadIdx.x;
    float w = W[k * DIM_EMBED + d];
    bf16_t hi = (bf16_t)w;
    Whi[k * DIM_EMBED + d] = hi;
    Wlo[k * DIM_EMBED + d] = (bf16_t)(w - (float)hi);
    red[d] = w * w;
    __syncthreads();
    for (int s = DIM_EMBED / 2; s > 0; s >>= 1) {
        if (d < s) red[d] += red[d + s];
        __syncthreads();
    }
    if (d == 0) wn2[k] = red[0];
}

// -------------------- kernel 1: WMMA distances + argmin --------------------
static __device__ inline v8f wmma_bf16(const v16bf a, const v16bf b, v8f c) {
    return __builtin_amdgcn_wmma_f32_16x16x32_bf16(
        /*neg_a=*/false, a, /*neg_b=*/false, b,
        /*c_mod=*/(short)0, c, /*reuse_a=*/false, /*reuse_b=*/false);
}

union Frag { v16bf v; v8bf h[2]; };

__global__ __launch_bounds__(THREADS)
void vq_argmin(const float* __restrict__ x, const bf16_t* __restrict__ Whi,
               const bf16_t* __restrict__ Wlo, const float* __restrict__ wn2,
               int* __restrict__ idx_ws, int* __restrict__ idx_out) {
    __shared__ __align__(16) bf16_t sAhi[M_TILE * A_STRIDE];
    __shared__ __align__(16) bf16_t sAlo[M_TILE * A_STRIDE];
    __shared__ __align__(16) bf16_t sBhi[KC * B_STRIDE];
    __shared__ __align__(16) bf16_t sBlo[KC * B_STRIDE];
    __shared__ __align__(16) float  sS[M_TILE * S_STRIDE];
    __shared__ float sWn[NUM_EMBED];

    const int tid = threadIdx.x;
    const int n0  = blockIdx.x * M_TILE;          // rows n0..n0+63, same batch b
    const int b   = n0 >> 11;                     // / T_SZ
    const int t0  = n0 & (T_SZ - 1);
    const float* xb = x + (size_t)b * DIM_EMBED * T_SZ;

    for (int i = tid; i < NUM_EMBED; i += THREADS) sWn[i] = wn2[i];

    // A tile: x[b, 0:256, t0:t0+64] -> transposed bf16 hi/lo [row=t][d]
    // 4096 float4 loads, coalesced along t.
    for (int it = 0; it < (M_TILE * DIM_EMBED / 4) / THREADS; ++it) {
        int i4  = it * THREADS + tid;
        int tt4 = (i4 & (M_TILE / 4 - 1)) * 4;
        int d   = i4 >> 4;                        // / (M_TILE/4)
        v4f v = *(const v4f*)(xb + (size_t)d * T_SZ + t0 + tt4);
#pragma unroll
        for (int j = 0; j < 4; ++j) {
            float f = v[j];
            bf16_t hi = (bf16_t)f;
            sAhi[(tt4 + j) * A_STRIDE + d] = hi;
            sAlo[(tt4 + j) * A_STRIDE + d] = (bf16_t)(f - (float)hi);
        }
    }

    const int lane   = tid & 31;
    const int l15    = lane & 15;
    const int hiHalf = lane >> 4;                 // 0 | 1
    const int waveId = tid >> 5;
    const int rw = waveId >> 1;                   // 0..3 : row group of 16
    const int wk = waveId & 1;                    // 0..1 : 32-col group

    const bf16_t* aHiRow = sAhi + (rw * 16 + l15) * A_STRIDE;
    const bf16_t* aLoRow = sAlo + (rw * 16 + l15) * A_STRIDE;
    const int kcol0 = wk * 32 + l15;              // frag 0 column (code)
    const int kcol1 = kcol0 + 16;                 // frag 1 column

    float bestV = 3.4e38f;
    int   bestI = 0;

    for (int kc = 0; kc < NUM_EMBED / KC; ++kc) {
        __syncthreads();
        // stage B chunk (64 codes x 256 dims, bf16 hi/lo) as u32 copies
        const unsigned* srcHi =
            (const unsigned*)(Whi + (size_t)kc * KC * DIM_EMBED);
        const unsigned* srcLo =
            (const unsigned*)(Wlo + (size_t)kc * KC * DIM_EMBED);
        for (int it = 0; it < (KC * DIM_EMBED / 2) / THREADS; ++it) {
            int i  = it * THREADS + tid;
            int kk = i >> 7, d2 = i & 127;        // 128 u32 per code row
            ((unsigned*)sBhi)[kk * (B_STRIDE / 2) + d2] = srcHi[i];
            ((unsigned*)sBlo)[kk * (B_STRIDE / 2) + d2] = srcLo[i];
        }
        if (kc + 1 < NUM_EMBED / KC)              // warm next chunk into caches
            __builtin_prefetch(Whi + (size_t)(kc + 1) * KC * DIM_EMBED +
                               (size_t)tid * 64, 0, 1);
        __syncthreads();

        v8f acc0 = {}, acc1 = {};
#pragma unroll
        for (int dstep = 0; dstep < DIM_EMBED / 32; ++dstep) {
            const int dk = dstep * 32;
            // A fragment (16x32 bf16): lanes 0-15 K=0..15 in v0-7,
            // lanes 16-31 K=16..31  -> two aligned ds_load_b128 each.
            const int aoff = dk + hiHalf * 16;
            Frag aH, aL;
            aH.h[0] = *(const v8bf*)(aHiRow + aoff);
            aH.h[1] = *(const v8bf*)(aHiRow + aoff + 8);
            aL.h[0] = *(const v8bf*)(aLoRow + aoff);
            aL.h[1] = *(const v8bf*)(aLoRow + aoff + 8);
            // B fragment (32x16 bf16): column = code = lane&15,
            // K(d) = dk + (lane<16 ? 0 : 16) + 0..15 contiguous.
            const int boff = dk + hiHalf * 16;
            const bf16_t* bH0 = sBhi + kcol0 * B_STRIDE + boff;
            const bf16_t* bL0 = sBlo + kcol0 * B_STRIDE + boff;
            const bf16_t* bH1 = sBhi + kcol1 * B_STRIDE + boff;
            const bf16_t* bL1 = sBlo + kcol1 * B_STRIDE + boff;
            Frag fH0, fL0, fH1, fL1;
            fH0.h[0] = *(const v8bf*)(bH0); fH0.h[1] = *(const v8bf*)(bH0 + 8);
            fL0.h[0] = *(const v8bf*)(bL0); fL0.h[1] = *(const v8bf*)(bL0 + 8);
            fH1.h[0] = *(const v8bf*)(bH1); fH1.h[1] = *(const v8bf*)(bH1 + 8);
            fL1.h[0] = *(const v8bf*)(bL1); fL1.h[1] = *(const v8bf*)(bL1 + 8);
            // bf16x3: hi*hi + hi*lo + lo*hi  (~fp32-accurate dot products)
            acc0 = wmma_bf16(aH.v, fH0.v, acc0);
            acc0 = wmma_bf16(aH.v, fL0.v, acc0);
            acc0 = wmma_bf16(aL.v, fH0.v, acc0);
            acc1 = wmma_bf16(aH.v, fH1.v, acc1);
            acc1 = wmma_bf16(aH.v, fL1.v, acc1);
            acc1 = wmma_bf16(aL.v, fH1.v, acc1);
        }

        // C layout: VGPR r, lane l -> M = r + (l<16?0:8), N = l&15
#pragma unroll
        for (int r = 0; r < 8; ++r) {
            int row = rw * 16 + r + hiHalf * 8;
            sS[row * S_STRIDE + wk * 32 + l15]      = acc0[r];
            sS[row * S_STRIDE + wk * 32 + 16 + l15] = acc1[r];
        }
        __syncthreads();

        if (tid < M_TILE) {
            const float* srow = sS + tid * S_STRIDE;
            const int kb = kc * KC;
            for (int c = 0; c < KC; ++c) {        // ascending k, strict <
                float v = sWn[kb + c] - 2.0f * srow[c];
                if (v < bestV) { bestV = v; bestI = kb + c; }
            }
        }
    }

    if (tid < M_TILE) {
        int n = n0 + tid;
        idx_ws[n]  = bestI;
        idx_out[n] = bestI;                       // int32 bits into output slot
    }
}

// -------------------- kernel 2: gather + loss partials ---------------------
__global__ __launch_bounds__(THREADS)
void vq_gather_loss(const float* __restrict__ x, const float* __restrict__ W,
                    const int* __restrict__ idx, float* __restrict__ out,
                    float* __restrict__ partials) {
    __shared__ float red[THREADS];
    const int i4 = blockIdx.x * THREADS + threadIdx.x;
    const int n  = i4 >> 6;                       // 64 float4 per row of 256
    const int d4 = (i4 & 63) << 2;
    const int k  = idx[n];
    v4f w  = *(const v4f*)(W + (size_t)k * DIM_EMBED + d4);
    v4f xv = *(const v4f*)(x + (size_t)i4 * 4);   // same flat index pairing
    *(v4f*)(out + (size_t)i4 * 4) = w;            // q_out flat == gathered flat
    v4f dv = w - xv;
    red[threadIdx.x] = dv[0]*dv[0] + dv[1]*dv[1] + dv[2]*dv[2] + dv[3]*dv[3];
    __syncthreads();
    for (int s = THREADS / 2; s > 0; s >>= 1) {
        if (threadIdx.x < s) red[threadIdx.x] += red[threadIdx.x + s];
        __syncthreads();
    }
    if (threadIdx.x == 0) partials[blockIdx.x] = red[0];
}

// -------------------- kernel 3: deterministic final reduce -----------------
__global__ __launch_bounds__(THREADS)
void vq_loss_final(const float* __restrict__ partials,
                   float* __restrict__ loss_out) {
    __shared__ float red[THREADS];
    float s = 0.f;
    for (int j = threadIdx.x; j < N_PART; j += THREADS) s += partials[j];
    red[threadIdx.x] = s;
    __syncthreads();
    for (int st = THREADS / 2; st > 0; st >>= 1) {
        if (threadIdx.x < st) red[threadIdx.x] += red[threadIdx.x + st];
        __syncthreads();
    }
    if (threadIdx.x == 0)
        loss_out[0] = COMMIT_LOSS * (red[0] / (float)Q_ELEMS);
}

// ---------------------------------------------------------------------------
extern "C" void kernel_launch(void* const* d_in, const int* in_sizes, int n_in,
                              void* d_out, int out_size, void* d_ws,
                              size_t ws_size, hipStream_t stream) {
    (void)in_sizes; (void)n_in; (void)out_size; (void)ws_size;
    const float* x = (const float*)d_in[0];   // [64, 256, 2048] fp32
    const float* W = (const float*)d_in[1];   // [512, 256] fp32
    float* out = (float*)d_out;               // q_out | loss | indices(int32)

    char* ws = (char*)d_ws;
    bf16_t* Whi     = (bf16_t*)(ws + WS_WHI);
    bf16_t* Wlo     = (bf16_t*)(ws + WS_WLO);
    float*  wn2     = (float*)(ws + WS_WN2);
    int*    idx     = (int*)(ws + WS_IDX);
    float*  parts   = (float*)(ws + WS_PART);
    int*    idx_out = (int*)(out + Q_ELEMS + 1);

    vq_prep_w<<<NUM_EMBED, DIM_EMBED, 0, stream>>>(W, Whi, Wlo, wn2);
    vq_argmin<<<N_ROWS / M_TILE, THREADS, 0, stream>>>(x, Whi, Wlo, wn2,
                                                       idx, idx_out);
    vq_gather_loss<<<Q_ELEMS / 4 / THREADS, THREADS, 0, stream>>>(x, W, idx,
                                                                  out, parts);
    vq_loss_final<<<1, THREADS, 0, stream>>>(parts, out + Q_ELEMS);
}
